// Trainer_41412074668213
// MI455X (gfx1250) — matrix operand order repaired and checked by reading, not compiled, not run on hardware
//
#include <hip/hip_runtime.h>
#include <hip/hip_bf16.h>

// ---------------------------------------------------------------------------
// LSTM encoder-decoder for MI455X (gfx1250, wave32, WMMA bf16).
//
//  K0: transpose/convert W_x, W_h -> bf16 [N][K] panels; also W_xh = W_x+W_h.
//  K1: z_x = (BN(pulse)@W_pe + b_pe) @ W_x + b_lstm   (big WMMA GEMM, no deps)
//  K2: persistent recurrence: 16 WGs (16 batch rows each), 16 waves/WG,
//      each wave owns one 16-col block of H and its 4 gate tiles.
//      - gate-i quarter of W_h (128 KB) staged in LDS (CDNA5 WGP has 320 KB)
//      - remaining B fragments double-buffered from L2 to hide latency
//      Encoder: z = z_x[t] + h@W_h  (32 wmma/wave/step)
//      Decoder step0: z = emb@W_x + h@W_h + b   (64 wmma/wave)
//      Decoder steps>=1: inp==h  =>  z = h@W_xh + b  (32 wmma/wave)
//      MLP head + scale fused at each decoder step (wave 0).
// ---------------------------------------------------------------------------

typedef __attribute__((ext_vector_type(16))) __bf16 v16bf;
typedef __attribute__((ext_vector_type(8)))  __bf16 v8bf;
typedef __attribute__((ext_vector_type(8)))  float  v8f;

#define B_   256
#define T_   512
#define F_   8
#define H_   256
#define H4_  1024
#define DEC_ 35
#define NT_  64   // 16x16 N-tiles across 4H
#define KT_  8    // K-fragments of 32 across H

__device__ __forceinline__ v8f wmma_bf16(v16bf a, v16bf b, v8f c) {
    return __builtin_amdgcn_wmma_f32_16x16x32_bf16(
        /*neg_a=*/false, a, /*neg_b=*/false, b,
        /*c_mod=*/(short)0, c, /*reuse_a=*/false, /*reuse_b=*/false);
}

// B fragment (32x16, bf16) from a [N][K(256)] panel (global or LDS):
// lanes 0-15 : col = tile*16+lane,    K = k0+{0..7, 16..23}
// lanes 16-31: col = tile*16+lane-16, K = k0+{8..15, 24..31}
__device__ __forceinline__ v16bf load_b_frag(const __bf16* Wt, int tile, int k0, int lane) {
    int n = tile * 16 + (lane & 15);
    int k = k0 + ((lane >> 4) << 3);
    union { v16bf v; v8bf h[2]; } u;
    u.h[0] = *(const v8bf*)(Wt + n * H_ + k);
    u.h[1] = *(const v8bf*)(Wt + n * H_ + k + 16);
    return u.v;
}

// A fragment (16x32, bf16) from a row-major [16][256] buffer (LDS):
// lanes 0-15 : row = lane,    K = k0+{0..7, 16..23}
// lanes 16-31: row = lane-16, K = k0+{8..15, 24..31}
__device__ __forceinline__ v16bf load_a_frag(const __bf16* hbuf, int k0, int lane) {
    int row = lane & 15;
    int kb  = k0 + ((lane >> 4) << 3);
    union { v16bf v; v8bf h[2]; } u;
    u.h[0] = *(const v8bf*)(hbuf + row * H_ + kb);
    u.h[1] = *(const v8bf*)(hbuf + row * H_ + kb + 16);
    return u.v;
}

__device__ __forceinline__ float sigf(float x)  { return 1.0f / (1.0f + __expf(-x)); }
__device__ __forceinline__ float tanhf_(float x){ float e = __expf(2.0f * x); return (e - 1.0f) / (e + 1.0f); }

// ---------------------------------------------------------------------------
// K0: weight convert + transpose. grid(1024) x block(256).
// ---------------------------------------------------------------------------
__global__ __launch_bounds__(256) void wconv_kernel(
    const float* __restrict__ Wx, const float* __restrict__ Wh,
    __bf16* __restrict__ wtx, __bf16* __restrict__ wth, __bf16* __restrict__ wtxh)
{
    int n = blockIdx.x;    // 0..1023 (output col of 4H)
    int k = threadIdx.x;   // 0..255  (input  row of H)
    float a = Wx[k * H4_ + n];
    float b = Wh[k * H4_ + n];
    wtx [n * H_ + k] = (__bf16)a;
    wth [n * H_ + k] = (__bf16)b;
    wtxh[n * H_ + k] = (__bf16)(a + b);
}

// ---------------------------------------------------------------------------
// K1: z_x = pe @ W_x + b_lstm, with pe recomputed per tile (BN + F=8 proj).
// grid(16*512) blocks, 256 threads (8 waves); wave w handles tiles w*8..w*8+7.
// z_x stored bf16 in per-lane WMMA C-fragment order for direct reload in K2.
// ---------------------------------------------------------------------------
__global__ __launch_bounds__(256) void zx_kernel(
    const float* __restrict__ pulse,
    const float* __restrict__ gamma, const float* __restrict__ beta,
    const float* __restrict__ mean,  const float* __restrict__ var,
    const float* __restrict__ Wpe,   const float* __restrict__ bpe,
    const float* __restrict__ b_lstm,
    const __bf16* __restrict__ wtx,  __bf16* __restrict__ zx)
{
    int blk = blockIdx.x >> 9;     // batch block 0..15
    int t   = blockIdx.x & 511;
    int tid = threadIdx.x;

    __shared__ float  bnx[16 * F_];
    __shared__ __bf16 pe[16 * H_];

    if (tid < 16 * F_) {
        int r = tid / F_, f = tid % F_;
        int b = blk * 16 + r;
        float x = pulse[((size_t)b * T_ + t) * F_ + f];
        bnx[tid] = (x - mean[f]) * rsqrtf(var[f] + 1e-3f) * gamma[f] + beta[f];
    }
    __syncthreads();

    {   // 16x256 pe tile: thread = column, loop rows (F=8 dot is trivial)
        int col = tid;
        float w[F_];
        #pragma unroll
        for (int f = 0; f < F_; ++f) w[f] = Wpe[f * H_ + col];
        float bp = bpe[col];
        #pragma unroll
        for (int r = 0; r < 16; ++r) {
            float acc = bp;
            #pragma unroll
            for (int f = 0; f < F_; ++f) acc += bnx[r * F_ + f] * w[f];
            pe[r * H_ + col] = (__bf16)acc;
        }
    }
    __syncthreads();

    int wave = tid >> 5, lane = tid & 31;
    for (int j = wave * 8; j < wave * 8 + 8; ++j) {
        float bb = b_lstm[j * 16 + (lane & 15)];
        v8f acc;
        #pragma unroll
        for (int r = 0; r < 8; ++r) acc[r] = bb;

        // double-buffered B fragments: hide L2 latency behind WMMA
        v16bf b = load_b_frag(wtx, j, 0, lane);
        #pragma unroll
        for (int kk = 0; kk < KT_; ++kk) {
            v16bf nb;
            if (kk + 1 < KT_) nb = load_b_frag(wtx, j, (kk + 1) * 32, lane);
            v16bf a = load_a_frag(pe, kk * 32, lane);
            acc = wmma_bf16(a, b, acc);
            if (kk + 1 < KT_) b = nb;
        }
        union { v8bf v; __bf16 e[8]; } st;
        #pragma unroll
        for (int r = 0; r < 8; ++r) st.e[r] = (__bf16)acc[r];
        *(v8bf*)(zx + ((size_t)(blk * T_ + t) * NT_ + j) * 256 + lane * 8) = st.v;
    }
}

// ---------------------------------------------------------------------------
// K2: persistent recurrence. grid(16) x block(512) = 16 waves.
// Wave jc owns H columns [jc*16, jc*16+16) and gate tiles {jc, jc+16, jc+32, jc+48}.
// h (16x256 bf16) in LDS; c in registers (C-fragment layout).
// Gate-i W_h panel (128 KB) staged in LDS once; other gates double-
// buffered from L2 (weights stay L2-resident: 192 MB L2 >> 1.5 MB weights).
// ---------------------------------------------------------------------------
__global__ __launch_bounds__(512) void rnn_kernel(
    const __bf16* __restrict__ zx,
    const __bf16* __restrict__ wth, const __bf16* __restrict__ wtx,
    const __bf16* __restrict__ wtxh,
    const float* __restrict__ b_lstm, const float* __restrict__ embed,
    const float* __restrict__ Wmlp,   const float* __restrict__ bmlp,
    const float* __restrict__ scw,    const float* __restrict__ scb,
    float* __restrict__ out)
{
    __shared__ __bf16 wi_lds[256 * H_];   // 128 KB: W_h panels n=0..255 (gate i)
    __shared__ __bf16 h_lds[16 * H_];     // 8 KB
    __shared__ __bf16 inp_lds[16 * H_];   // 8 KB

    int tid = threadIdx.x, lane = tid & 31, wave = tid >> 5;
    int blk = blockIdx.x;
    int jc  = wave;
    int tiles[4] = { jc, 16 + jc, 32 + jc, 48 + jc };

    // stage gate-i weight panels into LDS (amortized over 547 steps)
    for (int i = tid * 8; i < 256 * H_; i += 512 * 8)
        *(v8bf*)(wi_lds + i) = *(const v8bf*)(wth + i);
    for (int i = tid; i < 16 * H_; i += 512) h_lds[i] = (__bf16)0.0f;
    __syncthreads();

    v8f c;
    #pragma unroll
    for (int r = 0; r < 8; ++r) c[r] = 0.0f;
    float hreg[8];

    const __bf16* zx_blk = zx + (size_t)blk * T_ * NT_ * 256;

    // ---------------- encoder: 512 steps ----------------
    for (int t = 0; t < T_; ++t) {
        v8f acc[4];
        #pragma unroll
        for (int gi = 0; gi < 4; ++gi) {
            const __bf16* p = zx_blk + ((size_t)t * NT_ + tiles[gi]) * 256 + lane * 8;
            union { v8bf v; __bf16 e[8]; } ld;
            ld.v = *(const v8bf*)p;
            #pragma unroll
            for (int r = 0; r < 8; ++r) acc[gi][r] = (float)ld.e[r];
            if (t + 1 < T_) __builtin_prefetch(p + NT_ * 256, 0, 0);  // global_prefetch_b8
        }

        // double-buffered global B fragments (gates f,g,o); gate i from LDS
        v16bf b1 = load_b_frag(wth, tiles[1], 0, lane);
        v16bf b2 = load_b_frag(wth, tiles[2], 0, lane);
        v16bf b3 = load_b_frag(wth, tiles[3], 0, lane);
        #pragma unroll
        for (int kk = 0; kk < KT_; ++kk) {
            v16bf n1, n2, n3;
            if (kk + 1 < KT_) {
                n1 = load_b_frag(wth, tiles[1], (kk + 1) * 32, lane);
                n2 = load_b_frag(wth, tiles[2], (kk + 1) * 32, lane);
                n3 = load_b_frag(wth, tiles[3], (kk + 1) * 32, lane);
            }
            v16bf a  = load_a_frag(h_lds, kk * 32, lane);
            v16bf b0 = load_b_frag(wi_lds, jc, kk * 32, lane);   // LDS-resident
            acc[0] = wmma_bf16(a, b0, acc[0]);
            acc[1] = wmma_bf16(a, b1, acc[1]);
            acc[2] = wmma_bf16(a, b2, acc[2]);
            acc[3] = wmma_bf16(a, b3, acc[3]);
            if (kk + 1 < KT_) { b1 = n1; b2 = n2; b3 = n3; }
        }

        #pragma unroll
        for (int r = 0; r < 8; ++r) {
            float ig = sigf(acc[0][r]), fg = sigf(acc[1][r]);
            float gg = tanhf_(acc[2][r]), og = sigf(acc[3][r]);
            c[r]    = fg * c[r] + ig * gg;
            hreg[r] = og * tanhf_(c[r]);
        }
        __syncthreads();                       // all waves done reading h
        #pragma unroll
        for (int r = 0; r < 8; ++r) {
            int row = (lane < 16) ? r : r + 8;
            h_lds[row * H_ + jc * 16 + (lane & 15)] = (__bf16)hreg[r];
        }
        __syncthreads();                       // h updated for next step
    }

    // ---------------- decoder: 35 steps ----------------
    for (int i = tid; i < H_; i += 512) {
        __bf16 e = (__bf16)embed[i];
        #pragma unroll
        for (int r = 0; r < 16; ++r) inp_lds[r * H_ + i] = e;
    }
    __syncthreads();

    float sw0 = scw[0], sw1 = scw[1], sb0 = scb[0], sb1 = scb[1];
    float bm0 = bmlp[0], bm1 = bmlp[1];

    for (int s = 0; s < DEC_; ++s) {
        v8f acc[4];
        #pragma unroll
        for (int gi = 0; gi < 4; ++gi) {
            float bb = b_lstm[tiles[gi] * 16 + (lane & 15)];
            #pragma unroll
            for (int r = 0; r < 8; ++r) acc[gi][r] = bb;
        }
        if (s == 0) {
            // z = embed@W_x + h@W_h + b   (one step; no pipelining needed)
            #pragma unroll
            for (int kk = 0; kk < KT_; ++kk) {
                v16bf ax = load_a_frag(inp_lds, kk * 32, lane);
                v16bf ah = load_a_frag(h_lds,   kk * 32, lane);
                #pragma unroll
                for (int gi = 0; gi < 4; ++gi) {
                    acc[gi] = wmma_bf16(ax, load_b_frag(wtx, tiles[gi], kk * 32, lane), acc[gi]);
                    acc[gi] = wmma_bf16(ah, load_b_frag(wth, tiles[gi], kk * 32, lane), acc[gi]);
                }
            }
        } else {
            // inp == h  =>  z = h@(W_x + W_h) + b ; double-buffered B frags
            v16bf b0 = load_b_frag(wtxh, tiles[0], 0, lane);
            v16bf b1 = load_b_frag(wtxh, tiles[1], 0, lane);
            v16bf b2 = load_b_frag(wtxh, tiles[2], 0, lane);
            v16bf b3 = load_b_frag(wtxh, tiles[3], 0, lane);
            #pragma unroll
            for (int kk = 0; kk < KT_; ++kk) {
                v16bf n0, n1, n2, n3;
                if (kk + 1 < KT_) {
                    n0 = load_b_frag(wtxh, tiles[0], (kk + 1) * 32, lane);
                    n1 = load_b_frag(wtxh, tiles[1], (kk + 1) * 32, lane);
                    n2 = load_b_frag(wtxh, tiles[2], (kk + 1) * 32, lane);
                    n3 = load_b_frag(wtxh, tiles[3], (kk + 1) * 32, lane);
                }
                v16bf a = load_a_frag(h_lds, kk * 32, lane);
                acc[0] = wmma_bf16(a, b0, acc[0]);
                acc[1] = wmma_bf16(a, b1, acc[1]);
                acc[2] = wmma_bf16(a, b2, acc[2]);
                acc[3] = wmma_bf16(a, b3, acc[3]);
                if (kk + 1 < KT_) { b0 = n0; b1 = n1; b2 = n2; b3 = n3; }
            }
        }
        #pragma unroll
        for (int r = 0; r < 8; ++r) {
            float ig = sigf(acc[0][r]), fg = sigf(acc[1][r]);
            float gg = tanhf_(acc[2][r]), og = sigf(acc[3][r]);
            c[r]    = fg * c[r] + ig * gg;
            hreg[r] = og * tanhf_(c[r]);
        }
        __syncthreads();
        #pragma unroll
        for (int r = 0; r < 8; ++r) {
            int row = (lane < 16) ? r : r + 8;
            h_lds[row * H_ + jc * 16 + (lane & 15)] = (__bf16)hreg[r];
        }
        __syncthreads();

        // MLP head + scale: 16 rows x 2 outputs = 32 lanes of wave 0.
        // (next h_lds overwrite happens only after the next barrier pair)
        if (wave == 0) {
            int row = lane >> 1, cc = lane & 1;
            float d = 0.0f;
            for (int k = 0; k < H_; ++k)
                d += (float)h_lds[row * H_ + k] * Wmlp[k * 2 + cc];
            float v = (d + (cc ? bm1 : bm0)) * (cc ? sw1 : sw0) + (cc ? sb1 : sb0);
            out[(((size_t)blk * 16 + row) * DEC_ + s) * 2 + cc] = v;
        }
    }
}

// ---------------------------------------------------------------------------
extern "C" void kernel_launch(void* const* d_in, const int* in_sizes, int n_in,
                              void* d_out, int out_size, void* d_ws, size_t ws_size,
                              hipStream_t stream) {
    const float* pulse  = (const float*)d_in[0];
    const float* gamma  = (const float*)d_in[1];
    const float* beta   = (const float*)d_in[2];
    const float* mean   = (const float*)d_in[3];
    const float* var    = (const float*)d_in[4];
    const float* Wpe    = (const float*)d_in[5];
    const float* bpe    = (const float*)d_in[6];
    const float* embed  = (const float*)d_in[7];
    const float* Wx     = (const float*)d_in[8];
    const float* Wh     = (const float*)d_in[9];
    const float* b_lstm = (const float*)d_in[10];
    const float* Wmlp   = (const float*)d_in[11];
    const float* bmlp   = (const float*)d_in[12];
    const float* scw    = (const float*)d_in[13];
    const float* scb    = (const float*)d_in[14];

    char* ws = (char*)d_ws;
    __bf16* wtx  = (__bf16*)(ws + 0 * (size_t)(1 << 19));  // 512 KB
    __bf16* wth  = (__bf16*)(ws + 1 * (size_t)(1 << 19));  // 512 KB
    __bf16* wtxh = (__bf16*)(ws + 2 * (size_t)(1 << 19));  // 512 KB
    __bf16* zx   = (__bf16*)(ws + 3 * (size_t)(1 << 19));  // 256 MB

    wconv_kernel<<<dim3(H4_), dim3(H_), 0, stream>>>(Wx, Wh, wtx, wth, wtxh);
    zx_kernel<<<dim3((B_ / 16) * T_), dim3(256), 0, stream>>>(
        pulse, gamma, beta, mean, var, Wpe, bpe, b_lstm, wtx, zx);
    rnn_kernel<<<dim3(B_ / 16), dim3(512), 0, stream>>>(
        zx, wth, wtx, wtxh, b_lstm, embed, Wmlp, bmlp, scw, scb, (float*)d_out);
}
